// BreakthroughSNN_6210522710742
// MI455X (gfx1250) — matrix operand order repaired and checked by reading, not compiled, not run on hardware
//
#include <hip/hip_runtime.h>
#include <hip/hip_bf16.h>

typedef __attribute__((ext_vector_type(16))) __bf16 v16bf;
typedef __attribute__((ext_vector_type(8)))  __bf16 v8bf;
typedef __attribute__((ext_vector_type(8)))  float  v8f;

constexpr int B_ = 16, S_ = 128, D_ = 512, L_ = 3, V_ = 32000;
constexpr float DECAY = 0.6065306597126334f;   // exp(-1/2)
constexpr float THR   = 1.0f;
constexpr float EPS_  = 1e-5f;

#define DEV static __device__ __forceinline__

// ---------------- WMMA fragment helpers (contiguous k order, same for A and B) ----------------

// A-matrix 16x32 bf16: lanes 0-15 -> M=0..15, K 0..15; lanes 16-31 -> K 16..31
DEV v16bf fragA(const __bf16* base, int stride, int k0, int lane) {
  int m  = lane & 15;
  int kh = (lane >> 4) * 16;
  const __bf16* p = base + m * stride + k0 + kh;
  union { v16bf v; v8bf h[2]; } u;
  u.h[0] = *(const v8bf*)(p);
  u.h[1] = *(const v8bf*)(p + 8);
  return u.v;
}

// B-matrix 32x16 bf16 with B[k][n] = W[n][k], W row-major (n,k):
// lanes 0-15 -> n, K 0..15 (contiguous 32B); lanes 16-31 -> n, K 16..31
DEV v16bf fragB(const __bf16* W, int stride, int n0, int k0, int lane) {
  int n  = n0 + (lane & 15);
  int kk = k0 + ((lane >> 4) << 4);
  return *(const v16bf*)(W + n * stride + kk);
}

DEV v8f wmma_bf16(v16bf a, v16bf b, v8f c) {
  return __builtin_amdgcn_wmma_f32_16x16x32_bf16(
      /*neg_a=*/false, a, /*neg_b=*/false, b,
      /*c_mod=*/(short)0, c, /*reuse_a=*/false, /*reuse_b=*/false);
}

// ---------------- small GEMM: (16 x 512) @ W^T(512x512) + bias -> z (LDS f32) ----
// 16 waves, each computes two 16x16 N-tiles.  Weights streamed from L2 (bf16).
// unroll_count(2) keeps only ~4 B-fragments live -> no VGPR spill to scratch.
DEV void gemm16x512(const __bf16* aLds, const __bf16* __restrict__ W,
                    const float* __restrict__ bias, float* zLds,
                    int wave, int lane) {
  v8f acc0 = {}; v8f acc1 = {};
  const int n0 = wave * 32;
  const __bf16* wRow = W + (size_t)((n0 + (lane & 15)) * D_);
#pragma clang loop unroll_count(2)
  for (int k0 = 0; k0 < D_; k0 += 32) {
    __builtin_prefetch(wRow + k0 + 256, 0, 1);
    v16bf a  = fragA(aLds, D_, k0, lane);
    v16bf b0 = fragB(W, D_, n0,      k0, lane);
    acc0 = wmma_bf16(a, b0, acc0);
    v16bf b1 = fragB(W, D_, n0 + 16, k0, lane);
    acc1 = wmma_bf16(a, b1, acc1);
  }
  const int n  = lane & 15;
  const int mB = (lane >> 4) * 8;
#pragma unroll
  for (int v = 0; v < 8; ++v) {
    zLds[(mB + v) * D_ + n0 + n]      = acc0[v] + bias[n0 + n];
    zLds[(mB + v) * D_ + n0 + 16 + n] = acc1[v] + bias[n0 + 16 + n];
  }
}

// ---------------- LIF: mem = mem*decay + z; spike = (mem-1>=0); hard reset ----
// optional second destination (fused bottom-up copy)
DEV void lif_inplace(float* __restrict__ mem, float* z, float* z2, int tid) {
  for (int i = tid; i < B_ * D_; i += 512) {
    float m = mem[i] * DECAY + z[i];
    float s = (m - THR >= 0.f) ? 1.f : 0.f;
    mem[i] = m * (1.f - s);
    z[i]   = s;           // z now holds the spike
    if (z2) z2[i] = s;
  }
}

// ---------------- wave-per-row LayerNorm over D=512 -------------------------
// MODE 0: v = relu(x1 - x2)  -> bf16 outBf            (err = LN(relu(bu-pred)))
// MODE 1: v = x1 + x2        -> f32 outF, relu -> outF2 (new_state & next bu)
// MODE 2: v = relu(-x2)      -> bf16 outBf            (top-down err, bu = 0)
template <int MODE>
DEV void layernorm_row(const float* __restrict__ g, const float* __restrict__ b,
                       const float* x1, const float* x2,
                       float* outF, __bf16* outBf, float* outF2,
                       int wave, int lane) {
  float vbuf[16];
  float sum = 0.f;
  const int row = wave * D_;
#pragma unroll
  for (int i = 0; i < 16; ++i) {
    int d = lane + 32 * i;
    float v;
    if (MODE == 0)      v = fmaxf(x1[row + d] - x2[row + d], 0.f);
    else if (MODE == 1) v = x1[row + d] + x2[row + d];
    else                v = fmaxf(-x2[row + d], 0.f);
    vbuf[i] = v; sum += v;
  }
#pragma unroll
  for (int off = 16; off >= 1; off >>= 1) sum += __shfl_xor(sum, off, 32);
  float mu = sum * (1.f / D_);
  float var = 0.f;
#pragma unroll
  for (int i = 0; i < 16; ++i) { float dv = vbuf[i] - mu; var += dv * dv; }
#pragma unroll
  for (int off = 16; off >= 1; off >>= 1) var += __shfl_xor(var, off, 32);
  float rs = rsqrtf(var * (1.f / D_) + EPS_);
#pragma unroll
  for (int i = 0; i < 16; ++i) {
    int d = lane + 32 * i;
    float y = (vbuf[i] - mu) * rs * g[d] + b[d];
    if (MODE == 1) { outF[row + d] = y; outF2[row + d] = fmaxf(y, 0.f); }
    else           { outBf[row + d] = (__bf16)y; }
  }
}

// ---------------- fp32 -> bf16 conversion ------------------------------------
__global__ void cvt_bf16(const float* __restrict__ src, __bf16* __restrict__ dst, int n) {
  int i = blockIdx.x * blockDim.x + threadIdx.x;
  if (i < n) dst[i] = (__bf16)src[i];
}

// ---------------- recurrence: single resident workgroup, 128 serial steps ----
__global__ void __launch_bounds__(512)
snn_recurrence(const int* __restrict__ ids, const float* __restrict__ emb,
               const __bf16* __restrict__ encW, const float* __restrict__ encB,
               const __bf16* __restrict__ genW, const float* __restrict__ genB,
               const __bf16* __restrict__ infW, const float* __restrict__ infB,
               const float* __restrict__ nsG, const float* __restrict__ nsB,
               const float* __restrict__ neG, const float* __restrict__ neB,
               float* stWs, float* memEnc, float* memGen, float* memInf,
               __bf16* __restrict__ tdOut /* (B*S) x D, row = b*S + t */) {
  __shared__ __align__(32) float  zS [B_ * D_];   // 32 KB gemm out / spikes
  __shared__ __align__(32) float  buS[B_ * D_];   // 32 KB bottom-up signal
  __shared__ __align__(32) __bf16 aBf[B_ * D_];   // 16 KB WMMA A operand
  __shared__ __align__(32) __bf16 pBf[B_ * D_];   // 16 KB saved pred (td)

  const int tid = threadIdx.x, wave = tid >> 5, lane = tid & 31;

  // zero persistent state (deterministic per call)
  for (int i = tid; i < L_ * B_ * D_; i += 512) { stWs[i] = 0.f; memGen[i] = 0.f; memInf[i] = 0.f; }
  for (int i = tid; i < B_ * D_; i += 512) memEnc[i] = 0.f;
  __syncthreads();

  for (int t = 0; t < S_; ++t) {
    // embedding gather -> bf16 A
    for (int i = tid; i < B_ * D_; i += 512) {
      int b = i >> 9, d = i & 511;
      aBf[i] = (__bf16)emb[(size_t)ids[b * S_ + t] * D_ + d];
    }
    __syncthreads();
    gemm16x512(aBf, encW, encB, zS, wave, lane);
    __syncthreads();
    lif_inplace(memEnc, zS, buS, tid);            // zS = bu spikes, buS = copy
    __syncthreads();

    // ---- bottom-up pass ----
    for (int j = 0; j < L_; ++j) {
      float* stj = stWs + (size_t)j * B_ * D_;
      for (int i = tid; i < B_ * D_; i += 512) aBf[i] = (__bf16)stj[i];
      __syncthreads();
      gemm16x512(aBf, genW + (size_t)j * D_ * D_, genB + j * D_, zS, wave, lane);
      __syncthreads();
      lif_inplace(memGen + (size_t)j * B_ * D_, zS, nullptr, tid);  // zS = pred
      __syncthreads();
      layernorm_row<0>(neG + j * D_, neB + j * D_, buS, zS, nullptr, aBf, nullptr, wave, lane);
      __syncthreads();
      gemm16x512(aBf, infW + (size_t)j * D_ * D_, infB + j * D_, zS, wave, lane);
      __syncthreads();
      lif_inplace(memInf + (size_t)j * B_ * D_, zS, nullptr, tid);  // zS = su
      __syncthreads();
      layernorm_row<1>(nsG + j * D_, nsB + j * D_, stj, zS, stj, nullptr, buS, wave, lane);
      __syncthreads();
    }

    // ---- top-down pass (bu = 0, state discarded, mems updated) ----
    for (int i = tid; i < B_ * D_; i += 512) aBf[i] = (__bf16)stWs[2 * B_ * D_ + i];
    __syncthreads();
    for (int j = L_ - 1; j >= 0; --j) {
      gemm16x512(aBf, genW + (size_t)j * D_ * D_, genB + j * D_, zS, wave, lane);
      __syncthreads();
      lif_inplace(memGen + (size_t)j * B_ * D_, zS, nullptr, tid);  // zS = pred = new td
      __syncthreads();
      for (int i = tid; i < B_ * D_; i += 512) pBf[i] = (__bf16)zS[i];
      __syncthreads();
      layernorm_row<2>(neG + j * D_, neB + j * D_, nullptr, zS, nullptr, aBf, nullptr, wave, lane);
      __syncthreads();
      gemm16x512(aBf, infW + (size_t)j * D_ * D_, infB + j * D_, zS, wave, lane);
      __syncthreads();
      lif_inplace(memInf + (size_t)j * B_ * D_, zS, nullptr, tid);  // su discarded
      __syncthreads();
      for (int i = tid; i < B_ * D_; i += 512) aBf[i] = pBf[i];
      __syncthreads();
    }

    // stash td_t (bf16) for deferred logits GEMM, row = b*S + t
    for (int i = tid; i < B_ * D_; i += 512) {
      int b = i >> 9, d = i & 511;
      tdOut[((size_t)b * S_ + t) * D_ + d] = pBf[i];
    }
    __syncthreads();
  }
}

// ---------------- logits: (2048 x 512) @ outW^T(512 x 32000) + bias ----------
constexpr int BM = 128, BN = 128, BK = 64;
constexpr int LDA_ = BK + 8;    // 144B rows (16B aligned)
constexpr int LDB_ = BK + 16;   // 160B rows (32B aligned)

__global__ void __launch_bounds__(256)
logits_gemm(const __bf16* __restrict__ td, const __bf16* __restrict__ outW,
            const float* __restrict__ outB, float* __restrict__ out) {
  __shared__ __align__(32) __bf16 As[BM * LDA_];
  __shared__ __align__(32) __bf16 Bs[BN * LDB_];
  const int tid = threadIdx.x, wave = tid >> 5, lane = tid & 31;
  const int m0 = blockIdx.y * BM;
  const int n0 = blockIdx.x * BN;
  const int wm = (wave >> 2) * 64;   // 2x4 wave grid: 64x32 per wave
  const int wn = (wave & 3) * 32;

  v8f acc[4][2] = {};
  for (int k0 = 0; k0 < D_; k0 += BK) {
#pragma clang loop unroll_count(2)
    for (int i = tid; i < BM * (BK / 8); i += 256) {
      int r = i / (BK / 8), c = (i % (BK / 8)) * 8;
      *(v8bf*)&As[r * LDA_ + c] = *(const v8bf*)&td[(size_t)(m0 + r) * D_ + k0 + c];
    }
#pragma clang loop unroll_count(2)
    for (int i = tid; i < BN * (BK / 8); i += 256) {
      int r = i / (BK / 8), c = (i % (BK / 8)) * 8;
      const __bf16* gp = &outW[(size_t)(n0 + r) * D_ + k0 + c];
      *(v8bf*)&Bs[r * LDB_ + c] = *(const v8bf*)gp;
      if (k0 + BK < D_) __builtin_prefetch(gp + BK, 0, 1);
    }
    __syncthreads();
#pragma unroll
    for (int kk = 0; kk < BK; kk += 32) {
      v16bf b0 = fragB(Bs, LDB_, wn,      kk, lane);
      v16bf b1 = fragB(Bs, LDB_, wn + 16, kk, lane);
#pragma unroll
      for (int mi = 0; mi < 4; ++mi) {
        v16bf a = fragA(As + (wm + mi * 16) * LDA_, LDA_, kk, lane);
        acc[mi][0] = wmma_bf16(a, b0, acc[mi][0]);
        acc[mi][1] = wmma_bf16(a, b1, acc[mi][1]);
      }
    }
    __syncthreads();
  }
  const int nl = lane & 15, mB = (lane >> 4) * 8;
#pragma unroll
  for (int mi = 0; mi < 4; ++mi)
#pragma unroll
    for (int ni = 0; ni < 2; ++ni)
#pragma unroll
      for (int v = 0; v < 8; ++v) {
        int m = m0 + wm + mi * 16 + mB + v;
        int n = n0 + wn + ni * 16 + nl;
        out[(size_t)m * V_ + n] = acc[mi][ni][v] + outB[n];
      }
}

// ---------------- host launch -------------------------------------------------
extern "C" void kernel_launch(void* const* d_in, const int* in_sizes, int n_in,
                              void* d_out, int out_size, void* d_ws, size_t ws_size,
                              hipStream_t stream) {
  const int*   ids  = (const int*)  d_in[0];
  const float* emb  = (const float*)d_in[1];
  const float* encW = (const float*)d_in[2];
  const float* encB = (const float*)d_in[3];
  const float* genW = (const float*)d_in[4];
  const float* genB = (const float*)d_in[5];
  const float* infW = (const float*)d_in[6];
  const float* infB = (const float*)d_in[7];
  const float* nsG  = (const float*)d_in[8];
  const float* nsB  = (const float*)d_in[9];
  const float* neG  = (const float*)d_in[10];
  const float* neB  = (const float*)d_in[11];
  const float* outW = (const float*)d_in[12];
  const float* outB = (const float*)d_in[13];
  float* out = (float*)d_out;

  // workspace layout: bf16 region then 256B-aligned fp32 region
  __bf16* ws = (__bf16*)d_ws;
  size_t o = 0;
  __bf16* encBf  = ws + o; o += (size_t)D_ * D_;
  __bf16* genBf  = ws + o; o += (size_t)L_ * D_ * D_;
  __bf16* infBf  = ws + o; o += (size_t)L_ * D_ * D_;
  __bf16* outWBf = ws + o; o += (size_t)V_ * D_;
  __bf16* tdBf   = ws + o; o += (size_t)B_ * S_ * D_;
  size_t bytes = (o * 2 + 255) & ~(size_t)255;
  float* f = (float*)((char*)d_ws + bytes);
  float* stWs   = f; f += (size_t)L_ * B_ * D_;
  float* memEnc = f; f += (size_t)B_ * D_;
  float* memGen = f; f += (size_t)L_ * B_ * D_;
  float* memInf = f; f += (size_t)L_ * B_ * D_;

  // 1) convert weights to bf16 (L2-resident working set)
  auto cvt = [&](const float* s, __bf16* d, int n) {
    cvt_bf16<<<(n + 255) / 256, 256, 0, stream>>>(s, d, n);
  };
  cvt(encW, encBf, D_ * D_);
  cvt(genW, genBf, L_ * D_ * D_);
  cvt(infW, infBf, L_ * D_ * D_);
  cvt(outW, outWBf, V_ * D_);

  // 2) serial recurrence: one resident workgroup (16 waves) on one WGP
  snn_recurrence<<<1, 512, 0, stream>>>(ids, emb, encBf, encB, genBf, genB,
                                        infBf, infB, nsG, nsB, neG, neB,
                                        stWs, memEnc, memGen, memInf, tdBf);

  // 3) deferred logits projection: 2048 x 32000, bf16 WMMA
  dim3 grid(V_ / BN, (B_ * S_) / BM);
  logits_gemm<<<grid, 256, 0, stream>>>(tdBf, outWBf, outB, out);
}
